// Transformer_14568529068292
// MI455X (gfx1250) — matrix operand order repaired and checked by reading, not compiled, not run on hardware
//
#include <hip/hip_runtime.h>
#include <type_traits>

// ---------------------------------------------------------------------------
// CDNA5 (gfx1250) transformer forward.
//   - all matmuls via v_wmma_f32_16x16x32_bf16 (f32 accumulate)
//   - bf16 tiles staged with global_load_async_to_lds_b128 (ASYNCcnt)
//   - fp32 weight tiles staged via load + v_cvt_pk_bf16_f32
// ---------------------------------------------------------------------------

typedef __attribute__((ext_vector_type(16))) __bf16 v16bf;
typedef __attribute__((ext_vector_type(8)))  float  v8f;

constexpr int Lq = 2048;   // sequence length
constexpr int Dm = 1024;   // model dim
constexpr int Hh = 16;     // heads
constexpr int DH = 64;     // head dim
constexpr int Ff = 4096;   // ffn dim
constexpr int Vv_ = 32000; // vocab
constexpr int NLAYER = 4;
constexpr float EPSF = 1e-6f;

// ---------------------------------------------------------------------------
// Elementwise / normalization kernels
// ---------------------------------------------------------------------------
__global__ void embed_kernel(const int* __restrict__ ids,
                             const float* __restrict__ emb,
                             const float* __restrict__ pos,
                             float* __restrict__ x) {
  int idx = blockIdx.x * blockDim.x + threadIdx.x;   // over L*D
  int l = idx / Dm;
  int d = idx - l * Dm;
  x[idx] = emb[(size_t)ids[l] * Dm + d] + pos[idx];
}

__global__ void rmsnorm_bf16_kernel(const float* __restrict__ x,
                                    const float* __restrict__ scale,
                                    __bf16* __restrict__ out) {
  int row = blockIdx.x;
  const float* xr = x + (size_t)row * Dm;
  int t = threadIdx.x;
  float s = 0.f;
  for (int i = t; i < Dm; i += 256) { float v = xr[i]; s += v * v; }
  __shared__ float red[256];
  red[t] = s; __syncthreads();
  for (int o = 128; o > 0; o >>= 1) {
    if (t < o) red[t] += red[t + o];
    __syncthreads();
  }
  float inv = rsqrtf(red[0] / (float)Dm + EPSF);
  for (int i = t; i < Dm; i += 256)
    out[(size_t)row * Dm + i] = (__bf16)(xr[i] * inv * scale[i]);
}

__global__ void softmax_causal_kernel(const float* __restrict__ S,
                                      __bf16* __restrict__ P) {
  int row = blockIdx.x;
  const float* sr = S + (size_t)row * Lq;
  __bf16* pr = P + (size_t)row * Lq;
  int n = row + 1;                      // causal: columns 0..row valid
  int t = threadIdx.x;
  __shared__ float red[256];
  float m = -3.402823466e38f;
  for (int j = t; j < n; j += 256) m = fmaxf(m, sr[j]);
  red[t] = m; __syncthreads();
  for (int o = 128; o > 0; o >>= 1) {
    if (t < o) red[t] = fmaxf(red[t], red[t + o]);
    __syncthreads();
  }
  m = red[0]; __syncthreads();
  float s = 0.f;
  for (int j = t; j < n; j += 256) s += __expf(sr[j] - m);
  red[t] = s; __syncthreads();
  for (int o = 128; o > 0; o >>= 1) {
    if (t < o) red[t] += red[t + o];
    __syncthreads();
  }
  float inv = 1.f / red[0];
  for (int j = t; j < Lq; j += 256)
    pr[j] = (__bf16)((j < n) ? __expf(sr[j] - m) * inv : 0.f);
}

__global__ void swiglu_kernel(const float* __restrict__ G,
                              const float* __restrict__ Vx,
                              __bf16* __restrict__ Y, int n) {
  int i = blockIdx.x * blockDim.x + threadIdx.x;
  if (i < n) {
    float g = G[i];
    float y = (g / (1.f + __expf(-g))) * Vx[i];   // swish(g) * v
    Y[i] = (__bf16)y;
  }
}

__global__ void cast_bf16_kernel(const float* __restrict__ in,
                                 __bf16* __restrict__ out, int n) {
  int i = blockIdx.x * blockDim.x + threadIdx.x;
  if (i < n) out[i] = (__bf16)in[i];
}

// ---------------------------------------------------------------------------
// Async copy helper: one 16-byte memory->LDS transfer per lane (ASYNCcnt).
// lds byte offset = low 32 bits of the generic pointer (LDS aperture).
// ---------------------------------------------------------------------------
__device__ __forceinline__ void async_b128(unsigned lds_addr, const void* g) {
  asm volatile("global_load_async_to_lds_b128 %0, %1, off"
               :: "v"(lds_addr), "v"((unsigned long long)g)
               : "memory");
}
__device__ __forceinline__ void wait_async0() {
  asm volatile("s_wait_asynccnt 0x0" ::: "memory");
}

// ---------------------------------------------------------------------------
// Tiled WMMA GEMM:  C[M,N] (+)= alpha * A[M,K](bf16) * B[K,N]
//   TB      : element type of B (float weights or __bf16 activations)
//   TRANSB  : B stored as mem[n*ldb + k]
//   ADDC    : accumulate into f32 C (residual adds)
//   OUTBF16 : write bf16 C
//   CAUSAL  : skip tiles entirely above the diagonal (score GEMM)
// Block 128x64, BK=32, 128 threads = 4 waves (2x2); wave tile 64x32 = 4x2 WMMA.
// All dims are multiples of the tile sizes -> no bounds checks.
// ---------------------------------------------------------------------------
template <typename TB, bool TRANSB, bool ADDC, bool OUTBF16, bool CAUSAL>
__global__ __launch_bounds__(128) void gemm_wmma_kernel(
    const __bf16* __restrict__ A, const TB* __restrict__ B,
    void* __restrict__ Cp, int M, int N, int K,
    int lda, int ldb, int ldc, float alpha) {
  constexpr int BM = 128, BN = 64, BK = 32, STR = 48;  // 48-half stride (96B)
  __shared__ alignas(16) __bf16 As[BM * STR];
  __shared__ alignas(16) __bf16 Bs[BN * STR];          // stored [n][k]

  int m0 = blockIdx.y * BM, n0 = blockIdx.x * BN;
  if (CAUSAL && n0 > m0 + (BM - 1)) return;            // fully-masked tile

  int tid  = threadIdx.x;
  int lane = tid & 31;
  int wave = tid >> 5;
  int wr = wave >> 1, wc = wave & 1;

  unsigned as_base = (unsigned)(unsigned long long)(void*)As;
  unsigned bs_base = (unsigned)(unsigned long long)(void*)Bs;

  v8f c[4][2] = {};

  for (int k0 = 0; k0 < K; k0 += BK) {
    // ---- stage A tile (bf16 128x32): one row per thread, 4 async b128
    {
      const __bf16* src = A + (size_t)(m0 + tid) * lda + k0;
      unsigned dst = as_base + (unsigned)(tid * STR) * 2u;
      #pragma unroll
      for (int j = 0; j < 4; ++j)
        async_b128(dst + 16u * j, (const char*)src + 16 * j);
    }
    // ---- stage B tile (32x64 logical) into Bs[n][k]
    if constexpr (TRANSB && std::is_same<TB, __bf16>::value) {
      // bf16, k-contiguous rows: raw async copy (score GEMM K^T)
      int nr = tid >> 1, kb = (tid & 1) * 16;
      const __bf16* src = (const __bf16*)B + (size_t)(n0 + nr) * ldb + k0 + kb;
      unsigned dst = bs_base + (unsigned)(nr * STR + kb) * 2u;
      async_b128(dst,       src);
      async_b128(dst + 16u, src + 8);
    } else if constexpr (TRANSB) {
      int nr = tid >> 1, kb = (tid & 1) * 16;
      const TB* src = B + (size_t)(n0 + nr) * ldb + k0 + kb;
      #pragma unroll
      for (int j = 0; j < 16; ++j)
        Bs[nr * STR + kb + j] = (__bf16)(float)src[j];
      if (k0 + BK < K) __builtin_prefetch(src + BK, 0, 1);
    } else {
      int kr = tid >> 2, nb = (tid & 3) * 16;
      const TB* src = B + (size_t)(k0 + kr) * ldb + n0 + nb;
      #pragma unroll
      for (int j = 0; j < 16; ++j)
        Bs[(nb + j) * STR + kr] = (__bf16)(float)src[j];
      if (k0 + BK < K) __builtin_prefetch(src + (size_t)BK * ldb, 0, 1);
    }
    wait_async0();
    __syncthreads();

    // ---- fragments per ISA 16-bit layouts
    // A (16x32): row = lane%16;  K = kb..kb+7 and kb+16..kb+23, kb = lane<16?0:8
    int arow = wr * 64 + (lane & 15);
    int akb  = (lane < 16) ? 0 : 8;
    // B (32x16): col = lane%16;  K = kb..kb+7 and kb+16..kb+23, kb = lane<16?0:8
    int bcol = wc * 32 + (lane & 15);
    int bkb  = (lane < 16) ? 0 : 8;

    v16bf a[4], b[2];
    #pragma unroll
    for (int i = 0; i < 4; ++i) {
      const __bf16* p = &As[(arow + i * 16) * STR + akb];
      *(uint4*)&a[i]       = *(const uint4*)(p);
      *((uint4*)&a[i] + 1) = *(const uint4*)(p + 16);
    }
    #pragma unroll
    for (int j = 0; j < 2; ++j) {
      const __bf16* p = &Bs[(bcol + j * 16) * STR + bkb];
      *(uint4*)&b[j]       = *(const uint4*)(p);
      *((uint4*)&b[j] + 1) = *(const uint4*)(p + 16);
    }

    #pragma unroll
    for (int i = 0; i < 4; ++i)
      #pragma unroll
      for (int j = 0; j < 2; ++j)
        c[i][j] = __builtin_amdgcn_wmma_f32_16x16x32_bf16(
            false, a[i], false, b[j], (short)0, c[i][j], false, false);
    __syncthreads();
  }

  // ---- epilogue.  C frag: VGPR r -> M = r (+8 for lanes 16-31), N = lane%16
  int rofs = (lane < 16) ? 0 : 8;
  int ccol = lane & 15;
  #pragma unroll
  for (int i = 0; i < 4; ++i) {
    int rowb = m0 + wr * 64 + i * 16 + rofs;
    #pragma unroll
    for (int j = 0; j < 2; ++j) {
      int col = n0 + wc * 32 + j * 16 + ccol;
      #pragma unroll
      for (int r = 0; r < 8; ++r) {
        float val = alpha * c[i][j][r];
        size_t idx = (size_t)(rowb + r) * ldc + col;
        if (OUTBF16)       ((__bf16*)Cp)[idx] = (__bf16)val;
        else if (ADDC)     ((float*)Cp)[idx] += val;
        else               ((float*)Cp)[idx]  = val;
      }
    }
  }
}

// ---------------------------------------------------------------------------
// Host-side orchestration
// ---------------------------------------------------------------------------
extern "C" void kernel_launch(void* const* d_in, const int* in_sizes, int n_in,
                              void* d_out, int out_size, void* d_ws, size_t ws_size,
                              hipStream_t stream) {
  (void)in_sizes; (void)n_in; (void)out_size; (void)ws_size;

  const int*   ids    = (const int*)d_in[0];
  const float* embedw = (const float*)d_in[1];
  const float* pos    = (const float*)d_in[2];
  const float* wq     = (const float*)d_in[3];
  const float* wk     = (const float*)d_in[4];
  const float* wv     = (const float*)d_in[5];
  const float* wo     = (const float*)d_in[6];
  const float* gate_w = (const float*)d_in[7];
  const float* val_w  = (const float*)d_in[8];
  const float* out_w  = (const float*)d_in[9];
  const float* ln1    = (const float*)d_in[10];
  const float* ln2    = (const float*)d_in[11];
  const float* fln    = (const float*)d_in[12];
  float* logits = (float*)d_out;

  // Workspace carve-out (256B aligned)
  char* ws = (char*)d_ws;
  size_t off = 0;
  auto alloc = [&](size_t bytes) -> void* {
    off = (off + 255) & ~(size_t)255;
    void* p = ws + off;
    off += bytes;
    return p;
  };
  float*  x    = (float*) alloc((size_t)Lq * Dm * 4);
  __bf16* xn_a = (__bf16*)alloc((size_t)Lq * Dm * 2);   // ln1(x)
  __bf16* xn_f = (__bf16*)alloc((size_t)Lq * Dm * 2);   // ln2(x)
  __bf16* q    = (__bf16*)alloc((size_t)Lq * Dm * 2);
  __bf16* k    = (__bf16*)alloc((size_t)Lq * Dm * 2);
  __bf16* v    = (__bf16*)alloc((size_t)Lq * Dm * 2);
  float*  S    = (float*) alloc((size_t)Lq * Lq * 4);   // one head's scores
  __bf16* P    = (__bf16*)alloc((size_t)Lq * Lq * 2);
  float*  O    = (float*) alloc((size_t)Lq * Dm * 4);
  __bf16* Obf  = (__bf16*)alloc((size_t)Lq * Dm * 2);
  float*  G    = (float*) alloc((size_t)Lq * Ff * 4);
  float*  Vg   = (float*) alloc((size_t)Lq * Ff * 4);
  __bf16* Y    = (__bf16*)alloc((size_t)Lq * Ff * 2);

  dim3 blk(128);
  auto grid = [](int M, int N) { return dim3(N / 64, M / 128); };

  embed_kernel<<<(Lq * Dm) / 256, 256, 0, stream>>>(ids, embedw, pos, x);

  for (int i = 0; i < NLAYER; ++i) {
    const float* wq_i = wq + (size_t)i * Dm * Dm;
    const float* wk_i = wk + (size_t)i * Dm * Dm;
    const float* wv_i = wv + (size_t)i * Dm * Dm;
    const float* wo_i = wo + (size_t)i * Dm * Dm;
    const float* gw_i = gate_w + (size_t)i * Dm * Ff;
    const float* vw_i = val_w  + (size_t)i * Dm * Ff;
    const float* ow_i = out_w  + (size_t)i * Ff * Dm;

    // Both branch inputs from the SAME x (parallel residual)
    rmsnorm_bf16_kernel<<<Lq, 256, 0, stream>>>(x, ln1 + (size_t)i * Dm, xn_a);
    rmsnorm_bf16_kernel<<<Lq, 256, 0, stream>>>(x, ln2 + (size_t)i * Dm, xn_f);

    // QKV projections -> bf16
    gemm_wmma_kernel<float, false, false, true, false><<<grid(Lq, Dm), blk, 0, stream>>>(
        xn_a, wq_i, q, Lq, Dm, Dm, Dm, Dm, Dm, 1.f);
    gemm_wmma_kernel<float, false, false, true, false><<<grid(Lq, Dm), blk, 0, stream>>>(
        xn_a, wk_i, k, Lq, Dm, Dm, Dm, Dm, Dm, 1.f);
    gemm_wmma_kernel<float, false, false, true, false><<<grid(Lq, Dm), blk, 0, stream>>>(
        xn_a, wv_i, v, Lq, Dm, Dm, Dm, Dm, Dm, 1.f);

    // Attention, one head at a time (reuses the single S/P buffers)
    for (int h = 0; h < Hh; ++h) {
      // S = (Q_h K_h^T) / sqrt(dh), lower-triangular tiles only
      gemm_wmma_kernel<__bf16, true, false, false, true><<<grid(Lq, Lq), blk, 0, stream>>>(
          q + h * DH, k + h * DH, S, Lq, Lq, DH, Dm, Dm, Lq, 0.125f);
      softmax_causal_kernel<<<Lq, 256, 0, stream>>>(S, P);
      // O_h = P V_h  (writes column slice h*64..h*64+63 of O)
      gemm_wmma_kernel<__bf16, false, false, false, false><<<grid(Lq, DH), blk, 0, stream>>>(
          P, v + h * DH, O + h * DH, Lq, DH, Lq, Lq, Dm, Dm, 1.f);
    }
    cast_bf16_kernel<<<(Lq * Dm) / 256, 256, 0, stream>>>(O, Obf, Lq * Dm);
    // x += O @ wo
    gemm_wmma_kernel<float, false, true, false, false><<<grid(Lq, Dm), blk, 0, stream>>>(
        Obf, wo_i, x, Lq, Dm, Dm, Dm, Dm, Dm, 1.f);

    // FFN branch (from xn_f, computed before x was updated)
    gemm_wmma_kernel<float, false, false, false, false><<<grid(Lq, Ff), blk, 0, stream>>>(
        xn_f, gw_i, G, Lq, Ff, Dm, Dm, Ff, Ff, 1.f);
    gemm_wmma_kernel<float, false, false, false, false><<<grid(Lq, Ff), blk, 0, stream>>>(
        xn_f, vw_i, Vg, Lq, Ff, Dm, Dm, Ff, Ff, 1.f);
    swiglu_kernel<<<(Lq * Ff) / 256, 256, 0, stream>>>(G, Vg, Y, Lq * Ff);
    // x += Y @ ffn_out_w
    gemm_wmma_kernel<float, false, true, false, false><<<grid(Lq, Dm), blk, 0, stream>>>(
        Y, ow_i, x, Lq, Dm, Ff, Ff, Dm, Dm, 1.f);
  }

  // Final norm + tied-embedding logits: logits = xn @ embed^T
  rmsnorm_bf16_kernel<<<Lq, 256, 0, stream>>>(x, fln, xn_a);
  gemm_wmma_kernel<float, true, false, false, false><<<grid(Lq, Vv_), blk, 0, stream>>>(
      xn_a, embedw, logits, Lq, Vv_, Dm, Dm, Dm, Vv_, 1.f);
}
